// MultiHeadAttention_39178691674271
// MI455X (gfx1250) — compile-verified, matrix operand order checked by reading, not run on hardware
//
#include <hip/hip_runtime.h>

// ---------------------------------------------------------------------------
// MHA forward for gfx1250 (MI455X). bf16 WMMA, fp32 accumulate, all operands
// laid out so every fragment load is contiguous b128.
// d_out = [ out: B*S*D fp32 | attn_weights: B*H*S*S fp32 ]
// ---------------------------------------------------------------------------

typedef __attribute__((ext_vector_type(16))) __bf16 v16bf;
typedef __attribute__((ext_vector_type(8)))  __bf16 v8bf;
typedef __attribute__((ext_vector_type(8)))  float  v8f;
typedef __attribute__((ext_vector_type(4)))  float  f4;

static constexpr int B  = 2;
static constexpr int S  = 2048;
static constexpr int D  = 1024;
static constexpr int H  = 16;
static constexpr int DK = 64;
static constexpr long long M_ROWS    = (long long)B * S;      // 4096
static constexpr long long OUT_ELEMS = (long long)B * S * D;  // 4,194,304
static constexpr float SCALE = 0.125f;                        // dk^-0.5

// ---- WMMA ------------------------------------------------------------------

__device__ __forceinline__ v8f wmma_bf16(v16bf a, v16bf b, v8f c) {
  return __builtin_amdgcn_wmma_f32_16x16x32_bf16(
      false, a, false, b, (short)0, c, false, false);
}

// 16x32 bf16 fragment from a row-major [rows, ld] matrix. For B operands the
// source must be N-major (i.e. pre-transposed) so "row" == output column.
__device__ __forceinline__ v16bf ld_frag(const __bf16* __restrict__ base,
                                         int row0, int k0, int ld) {
  const int lane = threadIdx.x & 31;
  const __bf16* p =
      base + (long long)(row0 + (lane & 15)) * ld + (k0 + ((lane >> 4) << 3));
  const v8bf lo = *(const v8bf*)(p);
  const v8bf hi = *(const v8bf*)(p + 16);
  return __builtin_shufflevector(lo, hi, 0, 1, 2, 3, 4, 5, 6, 7,
                                 8, 9, 10, 11, 12, 13, 14, 15);
}

// Same fragment from fp32 source (softmax P), streamed non-temporally
// (last / only consumer), converted to bf16 on the fly.
__device__ __forceinline__ v16bf ld_frag_f32_nt(const float* __restrict__ base,
                                                int row0, int k0, int ld) {
  const int lane = threadIdx.x & 31;
  const float* p =
      base + (long long)(row0 + (lane & 15)) * ld + (k0 + ((lane >> 4) << 3));
  const f4 a0 = __builtin_nontemporal_load((const f4*)(p));
  const f4 a1 = __builtin_nontemporal_load((const f4*)(p + 4));
  const f4 b0 = __builtin_nontemporal_load((const f4*)(p + 16));
  const f4 b1 = __builtin_nontemporal_load((const f4*)(p + 20));
  v16bf r;
#pragma unroll
  for (int i = 0; i < 4; ++i) {
    r[i]      = (__bf16)a0[i];
    r[4 + i]  = (__bf16)a1[i];
    r[8 + i]  = (__bf16)b0[i];
    r[12 + i] = (__bf16)b1[i];
  }
  return r;
}

// ---- fp32 -> bf16 (vectorized, 8 elems/thread) ----------------------------

__global__ __launch_bounds__(256) void cvt_bf16(const float* __restrict__ in,
                                                __bf16* __restrict__ out) {
  const long long i = ((long long)blockIdx.x * 256 + threadIdx.x) * 8;
  const f4 a = *(const f4*)(in + i);
  const f4 b = *(const f4*)(in + i + 4);
  v8bf o;
#pragma unroll
  for (int j = 0; j < 4; ++j) { o[j] = (__bf16)a[j]; o[4 + j] = (__bf16)b[j]; }
  *(v8bf*)(out + i) = o;
}

// ---- fp32 [D,D] -> bf16 transposed [N=D, K=D] (LDS-tiled) -----------------

__global__ __launch_bounds__(256) void cvtT_bf16(const float* __restrict__ in,
                                                 __bf16* __restrict__ out) {
  __shared__ __bf16 tile[32][33];
  const int tx = threadIdx.x, ty = threadIdx.y;  // 32 x 8
  const int c = blockIdx.x * 32 + tx;
#pragma unroll
  for (int i = 0; i < 32; i += 8) {
    const int r = blockIdx.y * 32 + ty + i;
    tile[ty + i][tx] = (__bf16)in[(long long)r * D + c];
  }
  __syncthreads();
  const int r2 = blockIdx.y * 32 + tx;
#pragma unroll
  for (int i = 0; i < 32; i += 8) {
    const int c2 = blockIdx.x * 32 + ty + i;
    out[(long long)c2 * D + r2] = tile[tx][ty + i];
  }
}

// ---- Projection GEMM: 32x64 C-tile per wave, 2x4 register blocking ---------
// A: bf16 [M, D] row-major.  WT: bf16 [N=D, K=D] (pre-transposed).
// mode 0: RoPE, store bf16 [B,H,S,dk]      (Q, K)
// mode 1: store bf16 transposed [B,H,dk,S] (V, for contiguous PV B-fragments)
// mode 2: store fp32 [B,S,D]               (out-projection)

__device__ __forceinline__ void proj_epilogue(v8f acc, int n, int rbase,
                                              int mode,
                                              const float* __restrict__ bias,
                                              __bf16* __restrict__ outh,
                                              float* __restrict__ outf) {
  const float bn = bias[n];
  if (mode == 2) {
#pragma unroll
    for (int i = 0; i < 8; ++i)
      outf[(long long)(rbase + i) * D + n] = acc[i] + bn;
    return;
  }
  const int h = n >> 6;
  const int j = n & (DK - 1);
  float vals[8];
#pragma unroll
  for (int i = 0; i < 8; ++i) vals[i] = acc[i] + bn;

  if (mode == 0) {
    // Interleaved RoPE: columns n and n^1 live in lanes l and l^1.
    const int   half  = j >> 1;
    const float theta = __powf(10000.0f, -(float)(2 * half) * (1.0f / (float)DK));
#pragma unroll
    for (int i = 0; i < 8; ++i) {
      const int s = (rbase + i) & (S - 1);
      float sn, cs;
      __sincosf((float)s * theta, &sn, &cs);
      const float partner = __shfl_xor(vals[i], 1, 32);
      const float x2 = (j & 1) ? partner : -partner;
      vals[i] = vals[i] * cs + x2 * sn;
    }
#pragma unroll
    for (int i = 0; i < 8; ++i) {
      const int row = rbase + i;
      const int b = row >> 11, s = row & (S - 1);
      outh[(((long long)b * H + h) * S + s) * DK + j] = (__bf16)vals[i];
    }
  } else {  // mode 1: V transposed [B,H,dk,S]
#pragma unroll
    for (int i = 0; i < 8; ++i) {
      const int row = rbase + i;
      const int b = row >> 11, s = row & (S - 1);
      outh[(((long long)b * H + h) * DK + j) * S + s] = (__bf16)vals[i];
    }
  }
}

__global__ __launch_bounds__(256) void gemm_proj(
    const __bf16* __restrict__ A, const __bf16* __restrict__ WT,
    const float* __restrict__ bias, int mode,
    __bf16* __restrict__ outh, float* __restrict__ outf) {
  const int wave = threadIdx.x >> 5;
  const int lane = threadIdx.x & 31;
  const int wt = blockIdx.x * 8 + wave;
  const int nblk = D / 64;                 // 16
  const int m0 = (wt / nblk) * 32;
  const int n0 = (wt % nblk) * 64;

  v8f acc[2][4] = {};
  for (int k0 = 0; k0 < D; k0 += 32) {
    if (k0 + 32 < D)  // hint next K-slab of A into cache (global_prefetch_b8)
      __builtin_prefetch(A + (long long)(m0 + lane) * D + k0 + 32, 0, 3);
    const v16bf a0 = ld_frag(A, m0, k0, D);
    const v16bf a1 = ld_frag(A, m0 + 16, k0, D);
#pragma unroll
    for (int nj = 0; nj < 4; ++nj) {
      const v16bf b = ld_frag(WT, n0 + nj * 16, k0, D);
      acc[0][nj] = wmma_bf16(a0, b, acc[0][nj]);
      acc[1][nj] = wmma_bf16(a1, b, acc[1][nj]);
    }
  }
#pragma unroll
  for (int mi = 0; mi < 2; ++mi)
#pragma unroll
    for (int nj = 0; nj < 4; ++nj)
      proj_epilogue(acc[mi][nj], n0 + nj * 16 + (lane & 15),
                    m0 + mi * 16 + ((lane >> 4) << 3), mode, bias, outh, outf);
}

// ---- Scores: attn[bh,q,k] = scale * Q.K^T, 32x64 tile per wave -------------

__global__ __launch_bounds__(256) void attn_scores(
    const __bf16* __restrict__ Q, const __bf16* __restrict__ K,
    float* __restrict__ attn) {
  const int wave = threadIdx.x >> 5;
  const int lane = threadIdx.x & 31;
  const int wt = blockIdx.x * 8 + wave;
  const int kblk = S / 64;                 // 32
  const int per_bh = (S / 32) * kblk;      // 2048
  const int bh = wt / per_bh;
  const int r  = wt % per_bh;
  const int q0 = (r / kblk) * 32;
  const int c0 = (r % kblk) * 64;

  const __bf16* Qp = Q + (long long)bh * S * DK;
  const __bf16* Kp = K + (long long)bh * S * DK;

  v8f acc[2][4] = {};
#pragma unroll
  for (int k0 = 0; k0 < DK; k0 += 32) {
    const v16bf a0 = ld_frag(Qp, q0, k0, DK);
    const v16bf a1 = ld_frag(Qp, q0 + 16, k0, DK);
#pragma unroll
    for (int nj = 0; nj < 4; ++nj) {
      // B = K^T: K rows are contiguous along dk -> A-style fragment.
      const v16bf b = ld_frag(Kp, c0 + nj * 16, k0, DK);
      acc[0][nj] = wmma_bf16(a0, b, acc[0][nj]);
      acc[1][nj] = wmma_bf16(a1, b, acc[1][nj]);
    }
  }

  float* out = attn + (long long)bh * S * S;
#pragma unroll
  for (int mi = 0; mi < 2; ++mi)
#pragma unroll
    for (int nj = 0; nj < 4; ++nj) {
      const int n  = c0 + nj * 16 + (lane & 15);
      const int rb = q0 + mi * 16 + ((lane >> 4) << 3);
#pragma unroll
      for (int i = 0; i < 8; ++i)  // streamed: 537MB >> L2, bypass with NT
        __builtin_nontemporal_store(acc[mi][nj][i] * SCALE,
                                    &out[(long long)(rb + i) * S + n]);
    }
}

// ---- Online softmax: 2 reads + 1 write instead of 3R+2W --------------------

__global__ __launch_bounds__(256) void softmax_rows(float* __restrict__ attn) {
  float* p = attn + (long long)blockIdx.x * S;
  __shared__ float sm[256], ss[256];
  const int t = threadIdx.x;

  float m = -3.0e38f, sum = 0.0f;
  for (int i = t; i < S; i += 256) {
    const float v  = p[i];
    const float nm = fmaxf(m, v);
    sum = sum * __expf(m - nm) + __expf(v - nm);
    m = nm;
  }
  sm[t] = m; ss[t] = sum;
  __syncthreads();
  for (int s2 = 128; s2 > 0; s2 >>= 1) {
    if (t < s2) {
      const float m2 = sm[t + s2], v2 = ss[t + s2];
      const float nm = fmaxf(sm[t], m2);
      ss[t] = ss[t] * __expf(sm[t] - nm) + v2 * __expf(m2 - nm);
      sm[t] = nm;
    }
    __syncthreads();
  }
  const float M = sm[0], inv = 1.0f / ss[0];
  for (int i = t; i < S; i += 256) p[i] = __expf(p[i] - M) * inv;
}

// ---- O = P @ V : one wave covers 32 q-rows x all 64 head dims --------------

__global__ __launch_bounds__(256) void attn_pv(const float* __restrict__ attn,
                                               const __bf16* __restrict__ Vt,
                                               __bf16* __restrict__ O) {
  const int wave = threadIdx.x >> 5;
  const int lane = threadIdx.x & 31;
  const int wt = blockIdx.x * 8 + wave;     // < B*H*(S/32) = 2048
  const int per_bh = S / 32;                // 64
  const int bh = wt / per_bh;
  const int q0 = (wt % per_bh) * 32;

  const float*  P  = attn + (long long)bh * S * S;
  const __bf16* Vp = Vt + (long long)bh * DK * S;  // [dk, S]: N-major

  v8f acc[2][4] = {};
  for (int k0 = 0; k0 < S; k0 += 32) {
    const v16bf a0 = ld_frag_f32_nt(P, q0, k0, S);
    const v16bf a1 = ld_frag_f32_nt(P, q0 + 16, k0, S);
#pragma unroll
    for (int nj = 0; nj < 4; ++nj) {
      const v16bf b = ld_frag(Vp, nj * 16, k0, S);
      acc[0][nj] = wmma_bf16(a0, b, acc[0][nj]);
      acc[1][nj] = wmma_bf16(a1, b, acc[1][nj]);
    }
  }

  const int b_idx = bh >> 4;
  const int h     = bh & (H - 1);
#pragma unroll
  for (int mi = 0; mi < 2; ++mi)
#pragma unroll
    for (int nj = 0; nj < 4; ++nj) {
      const int j  = nj * 16 + (lane & 15);
      const int rb = q0 + mi * 16 + ((lane >> 4) << 3);
#pragma unroll
      for (int i = 0; i < 8; ++i) {
        const int s = rb + i;
        O[((long long)b_idx * S + s) * D + h * DK + j] =
            (__bf16)acc[mi][nj][i];
      }
    }
}

// ---------------------------------------------------------------------------

extern "C" void kernel_launch(void* const* d_in, const int* in_sizes, int n_in,
                              void* d_out, int out_size, void* d_ws,
                              size_t ws_size, hipStream_t stream) {
  (void)in_sizes; (void)n_in; (void)out_size; (void)ws_size;

  const float* x  = (const float*)d_in[0];
  const float* Wq = (const float*)d_in[1];
  const float* bq = (const float*)d_in[2];
  const float* Wk = (const float*)d_in[3];
  const float* bk = (const float*)d_in[4];
  const float* Wv = (const float*)d_in[5];
  const float* bv = (const float*)d_in[6];
  const float* Wo = (const float*)d_in[7];
  const float* bo = (const float*)d_in[8];

  float* out  = (float*)d_out;       // [B,S,D]
  float* attn = out + OUT_ELEMS;     // [B,H,S,S]

  char* ws = (char*)d_ws;            // ~48 MB total
  __bf16* xb   = (__bf16*)(ws);                  //  8 MB  x   [B*S, D]
  __bf16* WqT  = (__bf16*)(ws + (8ll  << 20));   //  2 MB  WqT [N, K]
  __bf16* WkT  = (__bf16*)(ws + (10ll << 20));   //  2 MB
  __bf16* WvT  = (__bf16*)(ws + (12ll << 20));   //  2 MB
  __bf16* WoT  = (__bf16*)(ws + (14ll << 20));   //  2 MB
  __bf16* Qr   = (__bf16*)(ws + (16ll << 20));   //  8 MB  Q+rope [B,H,S,dk]
  __bf16* Kr   = (__bf16*)(ws + (24ll << 20));   //  8 MB  K+rope
  __bf16* VtB  = (__bf16*)(ws + (32ll << 20));   //  8 MB  V^T [B,H,dk,S]
  __bf16* Ob   = (__bf16*)(ws + (40ll << 20));   //  8 MB  attn out [B,S,D]

  cvt_bf16<<<(int)(OUT_ELEMS / 8 / 256), 256, 0, stream>>>(x, xb);

  const dim3 tgrid(D / 32, D / 32), tblk(32, 8);
  cvtT_bf16<<<tgrid, tblk, 0, stream>>>(Wq, WqT);
  cvtT_bf16<<<tgrid, tblk, 0, stream>>>(Wk, WkT);
  cvtT_bf16<<<tgrid, tblk, 0, stream>>>(Wv, WvT);
  cvtT_bf16<<<tgrid, tblk, 0, stream>>>(Wo, WoT);

  const int projBlocks = (int)((M_ROWS / 32) * (D / 64) / 8);  // 256
  gemm_proj<<<projBlocks, 256, 0, stream>>>(xb, WqT, bq, 0, Qr, nullptr);
  gemm_proj<<<projBlocks, 256, 0, stream>>>(xb, WkT, bk, 0, Kr, nullptr);
  gemm_proj<<<projBlocks, 256, 0, stream>>>(xb, WvT, bv, 1, VtB, nullptr);

  const int scoreBlocks = B * H * (S / 32) * (S / 64) / 8;     // 8192
  attn_scores<<<scoreBlocks, 256, 0, stream>>>(Qr, Kr, attn);

  softmax_rows<<<B * H * S, 256, 0, stream>>>(attn);           // 65536 rows

  const int pvBlocks = B * H * (S / 32) / 8;                   // 256
  attn_pv<<<pvBlocks, 256, 0, stream>>>(attn, VtB, Ob);

  gemm_proj<<<projBlocks, 256, 0, stream>>>(Ob, WoT, bo, 2, nullptr, out);
}
